// MultiHeadAttentionBlock_82987358094082
// MI455X (gfx1250) — matrix operand order repaired and checked by reading, not compiled, not run on hardware
//
#include <hip/hip_runtime.h>
#include <hip/hip_bf16.h>

typedef _Float16 half_t;
typedef __attribute__((ext_vector_type(16))) _Float16 v16h;
typedef __attribute__((ext_vector_type(8)))  _Float16 v8h;
typedef __attribute__((ext_vector_type(4)))  _Float16 v4h;
typedef __attribute__((ext_vector_type(8)))  float    v8f;
typedef __attribute__((ext_vector_type(4)))  unsigned int u32x4;
typedef __attribute__((ext_vector_type(8)))  int i32x8;
typedef __attribute__((ext_vector_type(4)))  int i32x4;
typedef int v4i __attribute__((vector_size(16)));   // matches async builtin param type

#define D_MODEL 1024
#define HEADS   16
#define D_K     64
#define SEQ     2048
#define BATCH   2
#define MROWS   (BATCH * SEQ)

// ---------------- CDNA5 feature detection ----------------
#if __has_builtin(__builtin_amdgcn_tensor_load_to_lds)
#define CDNA5_HAS_TDM 1
#else
#define CDNA5_HAS_TDM 0
#endif
#if __has_include(<hip/amd_detail/amd_gfx1250_TDM.h>)
#define CDNA5_TDM_6ARG 1
#else
#define CDNA5_TDM_6ARG 0
#endif
#if __has_builtin(__builtin_amdgcn_global_load_async_to_lds_b128)
#define CDNA5_HAS_ASYNC 1
#else
#define CDNA5_HAS_ASYNC 0
#endif

static __device__ __forceinline__ v8f wmma_f16(v16h a, v16h b, v8f c) {
  return __builtin_amdgcn_wmma_f32_16x16x32_f16(false, a, false, b, (short)0, c,
                                                false, false);
}

// A/B fragment: halves 0..7 = first b128, halves 8..15 = second b128
static __device__ __forceinline__ v16h ld_frag(const half_t* p0, const half_t* p1) {
  v8h lo = *(const v8h*)p0;
  v8h hh = *(const v8h*)p1;
  v16h r;
#pragma unroll
  for (int i = 0; i < 8; ++i) { r[i] = lo[i]; r[i + 8] = hh[i]; }
  return r;
}

// async 16-byte global->LDS copy (per lane); falls back to sync copy
static __device__ __forceinline__ void cp_async16(half_t* l, const half_t* g) {
#if CDNA5_HAS_ASYNC
  __builtin_amdgcn_global_load_async_to_lds_b128((v4i*)g, (v4i*)l, 0, 0);
#else
  *(v8h*)l = *(const v8h*)g;
#endif
}
static __device__ __forceinline__ void wait_async0() {
#if CDNA5_HAS_ASYNC
#if __has_builtin(__builtin_amdgcn_s_wait_asynccnt)
  __builtin_amdgcn_s_wait_asynccnt(0);
#else
  asm volatile("s_wait_asynccnt 0x0" ::: "memory");
#endif
#endif
}

#if CDNA5_HAS_TDM
// Issue a 2D TDM load: tile (tile_d0 x tile_d1) f16 elements from a row-major
// tensor with row stride stride0 (elements) into LDS at lds_off (bytes).
static __device__ __forceinline__ void tdm_load_2d(uint32_t lds_off, const void* gaddr,
                                                   uint32_t tensor_d0, uint32_t tensor_d1,
                                                   uint32_t tile_d0, uint32_t tile_d1,
                                                   uint64_t stride0) {
  const uint64_t ga = (uint64_t)(uintptr_t)gaddr;
  u32x4 g0;
  g0[0] = 1u;                                   // count=1 (valid user descriptor)
  g0[1] = lds_off;                              // D#.lds_addr (bytes)
  g0[2] = (uint32_t)ga;                         // global_addr[31:0]
  g0[3] = (uint32_t)(ga >> 32) | (2u << 30);    // global_addr[56:32] | type=2
  i32x8 g1;
  g1[0] = (int)(1u << 16);                      // data_size=1 -> 2 bytes
  g1[1] = (int)((tensor_d0 & 0xFFFFu) << 16);   // tensor_dim0[15:0] @ bits 63:48
  g1[2] = (int)((tensor_d0 >> 16) | ((tensor_d1 & 0xFFFFu) << 16));
  g1[3] = (int)((tensor_d1 >> 16) | (tile_d0 << 16));   // tile_dim0
  g1[4] = (int)tile_d1;                          // tile_dim1 (tile_dim2=0)
  g1[5] = (int)(uint32_t)stride0;                // tensor_dim0_stride[31:0]
  g1[6] = (int)(uint32_t)(stride0 >> 32);        // stride[47:32], dim1_stride=0
  g1[7] = 0;
  i32x4 zz = {0, 0, 0, 0};
#if CDNA5_TDM_6ARG
  i32x8 z8 = {0, 0, 0, 0, 0, 0, 0, 0};
  __builtin_amdgcn_tensor_load_to_lds(g0, g1, zz, zz, z8, 0);
#else
  __builtin_amdgcn_tensor_load_to_lds(g0, g1, zz, zz, 0);
#endif
}
#endif

// ---------------- elementwise f32 -> f16 ----------------
__global__ __launch_bounds__(256) void cvt_f32_f16(const float* __restrict__ src,
                                                   half_t* __restrict__ dst, int n4) {
  int i = blockIdx.x * blockDim.x + threadIdx.x;
  if (i >= n4) return;
  float4 v = ((const float4*)src)[i];
  v4h o = {(half_t)v.x, (half_t)v.y, (half_t)v.z, (half_t)v.w};
  ((v4h*)dst)[i] = o;
}

// ------------- W[in][out] f32 -> Wt[out][in] f16 (tiled) -------------
__global__ __launch_bounds__(256) void cvt_w_transpose(const float* __restrict__ W,
                                                       half_t* __restrict__ Wt) {
  __shared__ float tile[32][33];
  const int tx = threadIdx.x, ty = threadIdx.y;
  const int bx = blockIdx.x * 32, by = blockIdx.y * 32;
#pragma unroll
  for (int j = 0; j < 32; j += 8)
    tile[ty + j][tx] = W[(size_t)(by + ty + j) * D_MODEL + bx + tx];
  __syncthreads();
#pragma unroll
  for (int j = 0; j < 32; j += 8)
    Wt[(size_t)(bx + ty + j) * D_MODEL + by + tx] = (half_t)tile[tx][ty + j];
}

// ------------- WMMA GEMM: Y = X(Mx1024) @ Wt^T + bias -------------
// Wt is [N][K] (K contiguous). mode 0: f16 head-major out; mode 1: f32 row-major.
// Tiles are DMA'd by the Tensor Data Mover with double buffering.
__global__ __launch_bounds__(256)
void gemm_wmma(const half_t* __restrict__ X, const half_t* __restrict__ Wt,
               const float* __restrict__ bias, half_t* __restrict__ out16,
               float* __restrict__ out32, int mode) {
  __shared__ alignas(16) half_t As[2][128 * 32];
  __shared__ alignas(16) half_t Bs[2][128 * 32];
  const int tid = threadIdx.x;
  const int w = tid >> 5, lane = tid & 31, l16 = lane & 15, hi = lane >> 4;
  const int bm = blockIdx.x * 128, bn = blockIdx.y * 128;

  v8f z = {0.f, 0.f, 0.f, 0.f, 0.f, 0.f, 0.f, 0.f};
  v8f acc[8];
#pragma unroll
  for (int t = 0; t < 8; ++t) acc[t] = z;

  const int NK = D_MODEL / 32;

#if CDNA5_HAS_TDM
  if (tid == 0) {
    tdm_load_2d((uint32_t)(uintptr_t)&As[0][0], X + (size_t)bm * D_MODEL,
                D_MODEL, MROWS, 32, 128, D_MODEL);
    tdm_load_2d((uint32_t)(uintptr_t)&Bs[0][0], Wt + (size_t)bn * D_MODEL,
                D_MODEL, D_MODEL, 32, 128, D_MODEL);
  }
#endif

  for (int ks = 0; ks < NK; ++ks) {
    const int cur = ks & 1;
#if CDNA5_HAS_TDM
    if (tid == 0) {
      if (ks + 1 < NK) {
        const int k1 = (ks + 1) * 32;
        tdm_load_2d((uint32_t)(uintptr_t)&As[cur ^ 1][0],
                    X + (size_t)bm * D_MODEL + k1, D_MODEL, MROWS, 32, 128, D_MODEL);
        tdm_load_2d((uint32_t)(uintptr_t)&Bs[cur ^ 1][0],
                    Wt + (size_t)bn * D_MODEL + k1, D_MODEL, D_MODEL, 32, 128, D_MODEL);
        __builtin_amdgcn_s_wait_tensorcnt(2);  // current stage's pair complete
      } else {
        __builtin_amdgcn_s_wait_tensorcnt(0);
      }
    }
    __syncthreads();
#else
    {  // fallback: per-lane async (or sync) copies, single buffered per stage
      const int k0 = ks * 32;
      const int row = tid >> 1, seg = tid & 1;
      const half_t* gx = X + (size_t)(bm + row) * D_MODEL + k0 + seg * 16;
      const half_t* gw = Wt + (size_t)(bn + row) * D_MODEL + k0 + seg * 16;
      cp_async16(&As[cur][row * 32 + seg * 16], gx);
      cp_async16(&As[cur][row * 32 + seg * 16 + 8], gx + 8);
      cp_async16(&Bs[cur][row * 32 + seg * 16], gw);
      cp_async16(&Bs[cur][row * 32 + seg * 16 + 8], gw + 8);
    }
    wait_async0();
    __syncthreads();
#endif

    const half_t* arow = &As[cur][(w * 16 + l16) * 32];
    v16h a = ld_frag(arow + hi * 8, arow + 16 + hi * 8);
    v16h bfr[8];
#pragma unroll
    for (int nt = 0; nt < 8; ++nt) {
      const half_t* brow = &Bs[cur][(nt * 16 + l16) * 32];
      bfr[nt] = ld_frag(brow + hi * 8, brow + 16 + hi * 8);
    }
#pragma unroll
    for (int nt = 0; nt < 8; ++nt) acc[nt] = wmma_f16(a, bfr[nt], acc[nt]);
    __syncthreads();
  }

#pragma unroll
  for (int nt = 0; nt < 8; ++nt) {
    const int c = bn + nt * 16 + l16;
    const float bv = bias[c];
#pragma unroll
    for (int r = 0; r < 8; ++r) {
      const int tok = bm + w * 16 + r + 8 * hi;
      const float val = acc[nt][r] + bv;
      if (mode == 0) {
        const int b = tok >> 11, s = tok & (SEQ - 1);
        const int h = c >> 6, d = c & (D_K - 1);
        out16[((size_t)((b * HEADS + h) * SEQ + s)) * D_K + d] = (half_t)val;
      } else {
        out32[(size_t)tok * D_MODEL + c] = val;
      }
    }
  }
}

// ------------- Flash attention: 64 q-rows per block, chunked keys -------------
__global__ __launch_bounds__(128)
void flash_attn(const half_t* __restrict__ Qh, const half_t* __restrict__ Kh,
                const half_t* __restrict__ Vh, const int* __restrict__ mask,
                half_t* __restrict__ ctx) {
  __shared__ alignas(16) half_t Qs[64 * 64];
  __shared__ alignas(16) half_t Ks[64 * 64];
  __shared__ alignas(16) half_t Vts[64 * 64];      // [d][key]
  __shared__ alignas(16) half_t Ps[4][16 * 64];    // per-wave P staging

  const int tid = threadIdx.x;
  const int w = tid >> 5, lane = tid & 31, l16 = lane & 15, hi = lane >> 4;
  const int qt = blockIdx.x, h = blockIdx.y, b = blockIdx.z;
  const int q0 = qt * 64;
  const size_t headBase = (size_t)(b * HEADS + h) * SEQ * D_K;

  {  // Q tile -> LDS via async copy (each wave loads the rows it consumes)
    const int row = tid >> 1, seg = tid & 1;
    const half_t* gq = Qh + headBase + (size_t)(q0 + row) * D_K + seg * 32;
    half_t* lq = &Qs[row * 64 + seg * 32];
#pragma unroll
    for (int i = 0; i < 4; ++i) cp_async16(lq + i * 8, gq + i * 8);
  }
  wait_async0();

  v8f z = {0.f, 0.f, 0.f, 0.f, 0.f, 0.f, 0.f, 0.f};
  v8f o[4];
#pragma unroll
  for (int t = 0; t < 4; ++t) o[t] = z;
  float rowmax[8], rowsum[8];
#pragma unroll
  for (int r = 0; r < 8; ++r) { rowmax[r] = -3.0e38f; rowsum[r] = 0.f; }

  for (int kc = 0; kc < SEQ / 64; ++kc) {
    const int k0 = kc * 64;
    __syncthreads();  // previous chunk consumers done before overwrite
    {
      const int row = tid >> 1, seg = tid & 1;
      const half_t* gk = Kh + headBase + (size_t)(k0 + row) * D_K + seg * 32;
      half_t* lk = &Ks[row * 64 + seg * 32];
#pragma unroll
      for (int i = 0; i < 4; ++i) cp_async16(lk + i * 8, gk + i * 8);
      const half_t* gv = Vh + headBase + (size_t)(k0 + row) * D_K + seg * 32;
#pragma unroll
      for (int i = 0; i < 4; ++i) {
        v8h vv = *(const v8h*)(gv + i * 8);
#pragma unroll
        for (int j = 0; j < 8; ++j)
          Vts[(seg * 32 + i * 8 + j) * 64 + row] = vv[j];  // transpose into LDS
      }
    }
    wait_async0();
    __syncthreads();

    // S = (Q @ K^T) for this wave's 16 rows x 64 keys
    v8f s4[4];
#pragma unroll
    for (int nt = 0; nt < 4; ++nt) s4[nt] = z;
#pragma unroll
    for (int ks = 0; ks < 2; ++ks) {
      const half_t* arow = &Qs[(w * 16 + l16) * 64 + ks * 32];
      v16h a = ld_frag(arow + hi * 8, arow + 16 + hi * 8);
#pragma unroll
      for (int nt = 0; nt < 4; ++nt) {
        const half_t* brow = &Ks[(nt * 16 + l16) * 64 + ks * 32];
        v16h bb = ld_frag(brow + hi * 8, brow + 16 + hi * 8);
        s4[nt] = wmma_f16(a, bb, s4[nt]);
      }
    }

    // scale + mask + online softmax (row = r + 8*hi, col lane = l16)
    float nmax[8];
#pragma unroll
    for (int r = 0; r < 8; ++r) nmax[r] = rowmax[r];
#pragma unroll
    for (int nt = 0; nt < 4; ++nt)
#pragma unroll
      for (int r = 0; r < 8; ++r) {
        float sv = s4[nt][r] * 0.125f;  // 1/sqrt(64)
        const int qrow = q0 + w * 16 + r + 8 * hi;
        const int kcol = k0 + nt * 16 + l16;
        const int mv = mask[((size_t)b * SEQ + qrow) * SEQ + kcol];
        sv = (mv != 0) ? sv : -1.0e9f;
        s4[nt][r] = sv;
        nmax[r] = fmaxf(nmax[r], sv);
      }
#pragma unroll
    for (int r = 0; r < 8; ++r)
#pragma unroll
      for (int off = 8; off >= 1; off >>= 1)
        nmax[r] = fmaxf(nmax[r], __shfl_xor(nmax[r], off, 16));

    float corr[8], rs[8];
#pragma unroll
    for (int r = 0; r < 8; ++r) {
      corr[r] = __expf(rowmax[r] - nmax[r]);
      rowmax[r] = nmax[r];
      rs[r] = 0.f;
    }
#pragma unroll
    for (int nt = 0; nt < 4; ++nt)
#pragma unroll
      for (int r = 0; r < 8; ++r) {
        float p = __expf(s4[nt][r] - nmax[r]);
        s4[nt][r] = p;
        rs[r] += p;
      }
#pragma unroll
    for (int r = 0; r < 8; ++r) {
#pragma unroll
      for (int off = 8; off >= 1; off >>= 1) rs[r] += __shfl_xor(rs[r], off, 16);
      rowsum[r] = rowsum[r] * corr[r] + rs[r];
    }
#pragma unroll
    for (int nt = 0; nt < 4; ++nt)
#pragma unroll
      for (int r = 0; r < 8; ++r) o[nt][r] *= corr[r];

    // restage P (C-layout -> A-layout) through wave-private LDS (same-wave
    // DS ops are in-order; no barrier required)
    half_t* pw = &Ps[w][0];
#pragma unroll
    for (int nt = 0; nt < 4; ++nt)
#pragma unroll
      for (int r = 0; r < 8; ++r)
        pw[(r + 8 * hi) * 64 + nt * 16 + l16] = (half_t)s4[nt][r];

    // O += P(16x64) @ V(64x64) via V^T staged in LDS
#pragma unroll
    for (int ks = 0; ks < 2; ++ks) {
      const half_t* arow = pw + l16 * 64 + ks * 32;
      v16h a = ld_frag(arow + hi * 8, arow + 16 + hi * 8);
#pragma unroll
      for (int nt = 0; nt < 4; ++nt) {
        const half_t* brow = &Vts[(nt * 16 + l16) * 64 + ks * 32];
        v16h bb = ld_frag(brow + hi * 8, brow + 16 + hi * 8);
        o[nt] = wmma_f16(a, bb, o[nt]);
      }
    }
  }

  // normalize + write context [B*S, D_MODEL] f16
#pragma unroll
  for (int r = 0; r < 8; ++r) {
    const float inv = 1.0f / rowsum[r];
    const int qrow = q0 + w * 16 + r + 8 * hi;
    const size_t base = ((size_t)b * SEQ + qrow) * D_MODEL + h * D_K;
#pragma unroll
    for (int nt = 0; nt < 4; ++nt)
      ctx[base + nt * 16 + l16] = (half_t)(o[nt][r] * inv);
  }
}

extern "C" void kernel_launch(void* const* d_in, const int* in_sizes, int n_in,
                              void* d_out, int out_size, void* d_ws, size_t ws_size,
                              hipStream_t stream) {
  (void)in_sizes; (void)n_in; (void)out_size; (void)ws_size;
  const float* q    = (const float*)d_in[0];
  const float* k    = (const float*)d_in[1];
  const float* v    = (const float*)d_in[2];
  const int*   mask = (const int*)d_in[3];
  const float* Wq = (const float*)d_in[4];
  const float* bq = (const float*)d_in[5];
  const float* Wk = (const float*)d_in[6];
  const float* bk = (const float*)d_in[7];
  const float* Wv = (const float*)d_in[8];
  const float* bv = (const float*)d_in[9];
  const float* Wo = (const float*)d_in[10];
  const float* bo = (const float*)d_in[11];
  float* out = (float*)d_out;

  half_t* ws = (half_t*)d_ws;
  const size_t ACT = (size_t)BATCH * SEQ * D_MODEL;  // 4M halves
  const size_t WEL = (size_t)D_MODEL * D_MODEL;      // 1M halves
  half_t* Xq  = ws;         half_t* Xk  = Xq + ACT;  half_t* Xv  = Xk + ACT;
  half_t* Wqt = Xv + ACT;   half_t* Wkt = Wqt + WEL;
  half_t* Wvt = Wkt + WEL;  half_t* Wot = Wvt + WEL;
  half_t* Qh  = Wot + WEL;  half_t* Kh  = Qh + ACT;  half_t* Vh = Kh + ACT;
  half_t* Ctx = Vh + ACT;   // total 32M halves = 64 MB

  {
    int n4 = (int)(ACT / 4);
    dim3 g((n4 + 255) / 256), blk(256);
    cvt_f32_f16<<<g, blk, 0, stream>>>(q, Xq, n4);
    cvt_f32_f16<<<g, blk, 0, stream>>>(k, Xk, n4);
    cvt_f32_f16<<<g, blk, 0, stream>>>(v, Xv, n4);
  }
  {
    dim3 g(D_MODEL / 32, D_MODEL / 32), blk(32, 8);
    cvt_w_transpose<<<g, blk, 0, stream>>>(Wq, Wqt);
    cvt_w_transpose<<<g, blk, 0, stream>>>(Wk, Wkt);
    cvt_w_transpose<<<g, blk, 0, stream>>>(Wv, Wvt);
    cvt_w_transpose<<<g, blk, 0, stream>>>(Wo, Wot);
  }
  {
    dim3 g(MROWS / 128, D_MODEL / 128), blk(256);
    gemm_wmma<<<g, blk, 0, stream>>>(Xq, Wqt, bq, Qh, nullptr, 0);
    gemm_wmma<<<g, blk, 0, stream>>>(Xk, Wkt, bk, Kh, nullptr, 0);
    gemm_wmma<<<g, blk, 0, stream>>>(Xv, Wvt, bv, Vh, nullptr, 0);
  }
  {
    dim3 g(SEQ / 64, HEADS, BATCH), blk(128);
    flash_attn<<<g, blk, 0, stream>>>(Qh, Kh, Vh, mask, Ctx);
  }
  {
    dim3 g(MROWS / 128, D_MODEL / 128), blk(256);
    gemm_wmma<<<g, blk, 0, stream>>>(Ctx, Wot, bo, nullptr, out, 1);
  }
}
